// SemanticMemory_69741678953013
// MI455X (gfx1250) — matrix operand order repaired and checked by reading, not compiled, not run on hardware
//
#include <hip/hip_runtime.h>
#include <hip/hip_bf16.h>

// ---------------------------------------------------------------------------
// Problem constants (from reference): B=2048, N=131072, H=512, TOP_K=4
// ---------------------------------------------------------------------------
#define BQ     2048
#define NKEYS  131072
#define HDIM   512
#define NCHUNK 32              // N split into 32 chunks of 4096 for sim/topk
#define CHUNKC 4096

typedef __bf16  v16bf __attribute__((ext_vector_type(16)));
typedef float   v8f   __attribute__((ext_vector_type(8)));

union FragBF {
    v16bf v;
    uint4 u[2];
};

__device__ __forceinline__ unsigned short f32_to_bf16_rne(float f) {
    unsigned int u = __float_as_uint(f);
    unsigned int r = u + 0x7FFFu + ((u >> 16) & 1u);
    return (unsigned short)(r >> 16);
}

// WGP-scope prefetch of 8 consecutive 128B cachelines (1 KB).
// Default (unannotated) scope == SCOPE 0 (WGP): fills all cache levels
// on a miss (ISA 10.5) -- unlike __builtin_prefetch, which maps to SE/SYS
// scope and never fills the WGP near-cache.
__device__ __forceinline__ void prefetch_row_1k_wgp(const void* p) {
    unsigned long long a = (unsigned long long)p;
    asm volatile(
        "global_prefetch_b8 %0, off\n\t"
        "global_prefetch_b8 %0, off offset:128\n\t"
        "global_prefetch_b8 %0, off offset:256\n\t"
        "global_prefetch_b8 %0, off offset:384\n\t"
        "global_prefetch_b8 %0, off offset:512\n\t"
        "global_prefetch_b8 %0, off offset:640\n\t"
        "global_prefetch_b8 %0, off offset:768\n\t"
        "global_prefetch_b8 %0, off offset:896"
        :: "v"(a) : "memory");
}

__device__ __forceinline__ void top4_insert(float v, int id,
                                            float& t0, float& t1, float& t2, float& t3,
                                            int& i0, int& i1, int& i2, int& i3) {
    if (v <= t3) return;
    if (v > t0)      { t3=t2; i3=i2; t2=t1; i2=i1; t1=t0; i1=i0; t0=v; i0=id; }
    else if (v > t1) { t3=t2; i3=i2; t2=t1; i2=i1; t1=v;  i1=id; }
    else if (v > t2) { t3=t2; i3=i2; t2=v;  i2=id; }
    else             { t3=v;  i3=id; }
}

// ---------------------------------------------------------------------------
// Kernel 0: f32 -> bf16 conversion (weights)
// ---------------------------------------------------------------------------
__global__ void cvt_bf16_kernel(const float* __restrict__ src,
                                unsigned short* __restrict__ dst, int n) {
    int i = blockIdx.x * blockDim.x + threadIdx.x;
    if (i < n) dst[i] = f32_to_bf16_rne(src[i]);
}

// ---------------------------------------------------------------------------
// Kernel 1/2/5: out[m][n] = sum_k X[m][k] * W[n][k]   (i.e. X @ W.T)
// Block: 16 rows x 512 cols, 128 threads = 4 waves, each wave 128 cols.
// NORM=true : L2-normalize each output row, store bf16 (for Qn / Kn).
// NORM=false: store f32 directly (final output projection).
// WMMA bf16 16x16x32; A tile staged in LDS as bf16; B explicitly
// double-buffered so loads overlap WMMA (partial loadcnt waits).
// ---------------------------------------------------------------------------
template <bool NORM>
__global__ __launch_bounds__(128)
void gemm_xwT_kernel(const float* __restrict__ X,
                     const unsigned short* __restrict__ Wbf,
                     void* __restrict__ out) {
    __shared__ unsigned short Albs[16 * HDIM];  // 16 KB bf16 A tile
    __shared__ float Cout[16 * HDIM];           // 32 KB f32 result tile
    __shared__ float rscale[16];

    const int tid     = threadIdx.x;
    const int rowbase = blockIdx.x * 16;

    // ---- stage + convert A tile: 16x512 f32 -> bf16 in LDS ----
    {
        const float* Xt = X + (size_t)rowbase * HDIM;
        unsigned int* Au = (unsigned int*)Albs;
        #pragma unroll
        for (int i = 0; i < 32; ++i) {            // 4096 packed uints / 128 thr
            int u  = i * 128 + tid;
            int e  = u * 2;
            unsigned short lo = f32_to_bf16_rne(Xt[e]);
            unsigned short hi = f32_to_bf16_rne(Xt[e + 1]);
            Au[u] = (unsigned int)lo | ((unsigned int)hi << 16);
        }
    }
    __syncthreads();

    const int wave = tid >> 5;
    const int lane = tid & 31;
    const int hlf  = lane >> 4;     // 0: lanes 0-15, 1: lanes 16-31
    const int lcol = lane & 15;

    // B frag loader: column n = wave*128 + t*16 + lcol, K = ks*32 + hlf*16 ..+15
    auto bload = [&](int t, int ks) -> FragBF {
        FragBF b;
        const unsigned short* bp =
            &Wbf[(size_t)(wave * 128 + t * 16 + lcol) * HDIM + ks * 32 + hlf * 16];
        b.u[0] = *(const uint4*)(bp);
        b.u[1] = *(const uint4*)(bp + 8);
        return b;
    };

    FragBF bc = bload(0, 0);        // pipeline prologue
    for (int t = 0; t < 8; ++t) {
        const int nbase = wave * 128 + t * 16;    // local output column base
        v8f acc = {0.f, 0.f, 0.f, 0.f, 0.f, 0.f, 0.f, 0.f};
        #pragma unroll
        for (int ks = 0; ks < 16; ++ks) {
            // issue next fragment's loads BEFORE consuming current fragment
            const int nt  = (ks == 15) ? ((t + 1) & 7) : t;   // wrap: harmless in-range load
            const int nks = (ks == 15) ? 0 : ks + 1;
            FragBF bn = bload(nt, nks);

            FragBF a;
            const unsigned short* ap = &Albs[lcol * HDIM + ks * 32 + hlf * 16];
            a.u[0] = *(const uint4*)(ap);
            a.u[1] = *(const uint4*)(ap + 8);

            acc = __builtin_amdgcn_wmma_f32_16x16x32_bf16(
                false, a.v, false, bc.v, (short)0, acc, false, false);
            bc = bn;
        }
        if (NORM) {
            #pragma unroll
            for (int r = 0; r < 8; ++r)
                Cout[(r + hlf * 8) * HDIM + nbase + lcol] = acc[r];
        } else {
            float* O = (float*)out;
            #pragma unroll
            for (int r = 0; r < 8; ++r)
                O[(size_t)(rowbase + r + hlf * 8) * HDIM + nbase + lcol] = acc[r];
        }
    }

    if (NORM) {
        __syncthreads();
        if (tid < 16) {
            float s = 0.f;
            const float* row = &Cout[tid * HDIM];
            for (int h = 0; h < HDIM; ++h) { float v = row[h]; s += v * v; }
            rscale[tid] = 1.0f / fmaxf(sqrtf(s), 1e-12f);
        }
        __syncthreads();
        unsigned short* O = (unsigned short*)out + (size_t)rowbase * HDIM;
        #pragma unroll
        for (int i = 0; i < 64; ++i) {
            int e = i * 128 + tid;
            int r = e >> 9, h = e & 511;
            O[e] = f32_to_bf16_rne(Cout[r * HDIM + h] * rscale[r]);
        }
    }
}

// ---------------------------------------------------------------------------
// Kernel 3: fused  sim = Qn @ Kn.T  +  running per-row top-4.
// Grid: (BQ/16 row-blocks, NCHUNK chunks). Block: 128 threads = 4 waves.
// Each wave: 16 rows x 1024 key-columns = 64 WMMA column tiles x 16 K-steps.
// B double-buffered in registers (loads overlap WMMA) + WGP-scope
// global_prefetch of the tile two steps ahead (Kn is L2-resident; the
// prefetch pulls the next tile's lines into the WGP near-cache so the
// register pipeline only covers L0-hit latency).
// Per-lane top-4 kept in registers, merged through LDS -> partial per chunk.
// ---------------------------------------------------------------------------
__global__ __launch_bounds__(128)
void sim_topk_kernel(const unsigned short* __restrict__ Qn,
                     const unsigned short* __restrict__ Kn,
                     float* __restrict__ pv, int* __restrict__ pi) {
    __shared__ unsigned short Albs[16 * HDIM];   // 16 KB Qn tile
    __shared__ float cvls[4][32][8][4];          // 16 KB candidate vals
    __shared__ int   cils[4][32][8][4];          // 16 KB candidate idx

    const int tid  = threadIdx.x;
    const int wave = tid >> 5;
    const int lane = tid & 31;
    const int hlf  = lane >> 4;
    const int lcol = lane & 15;
    const int rowbase   = blockIdx.x * 16;
    const int chunkbase = blockIdx.y * CHUNKC;

    // stage Qn tile (already bf16, just copy 16 KB)
    {
        const unsigned int* src = (const unsigned int*)(Qn + (size_t)rowbase * HDIM);
        unsigned int* dst = (unsigned int*)Albs;
        #pragma unroll
        for (int i = 0; i < 32; ++i) dst[i * 128 + tid] = src[i * 128 + tid];
    }
    __syncthreads();

    float tv[8][4];
    int   ti[8][4];
    #pragma unroll
    for (int r = 0; r < 8; ++r)
        #pragma unroll
        for (int s = 0; s < 4; ++s) { tv[r][s] = -3.0e38f; ti[r][s] = 0; }

    // B frag loader: key n = chunkbase + wave*1024 + t*16 + lcol
    auto bload = [&](int t, int ks) -> FragBF {
        FragBF b;
        const unsigned short* bp =
            &Kn[(size_t)(chunkbase + wave * 1024 + t * 16 + lcol) * HDIM + ks * 32 + hlf * 16];
        b.u[0] = *(const uint4*)(bp);
        b.u[1] = *(const uint4*)(bp + 8);
        return b;
    };

    FragBF bc = bload(0, 0);        // pipeline prologue
    for (int t = 0; t < 64; ++t) {
        // WGP-scope prefetch of this lane's Kn row for tile t+2 (1 KB = 8 lines)
        if (t + 2 < 64) {
            prefetch_row_1k_wgp(
                &Kn[(size_t)(chunkbase + wave * 1024 + (t + 2) * 16 + lcol) * HDIM]);
        }

        v8f acc = {0.f, 0.f, 0.f, 0.f, 0.f, 0.f, 0.f, 0.f};
        #pragma unroll
        for (int ks = 0; ks < 16; ++ks) {
            const int nt  = (ks == 15) ? ((t + 1) & 63) : t;  // wrap: in-range
            const int nks = (ks == 15) ? 0 : ks + 1;
            FragBF bn = bload(nt, nks);

            FragBF a;
            const unsigned short* ap = &Albs[lcol * HDIM + ks * 32 + hlf * 16];
            a.u[0] = *(const uint4*)(ap);
            a.u[1] = *(const uint4*)(ap + 8);

            acc = __builtin_amdgcn_wmma_f32_16x16x32_bf16(
                false, a.v, false, bc.v, (short)0, acc, false, false);
            bc = bn;
        }

        const int col = chunkbase + wave * 1024 + t * 16 + lcol;
        #pragma unroll
        for (int r = 0; r < 8; ++r) {
            top4_insert(acc[r], col,
                        tv[r][0], tv[r][1], tv[r][2], tv[r][3],
                        ti[r][0], ti[r][1], ti[r][2], ti[r][3]);
        }
    }

    // dump per-lane candidates
    #pragma unroll
    for (int r = 0; r < 8; ++r)
        #pragma unroll
        for (int s = 0; s < 4; ++s) {
            cvls[wave][lane][r][s] = tv[r][s];
            cils[wave][lane][r][s] = ti[r][s];
        }
    __syncthreads();

    // lanes 0..15 of the block: each merges one row (256 candidates -> top4)
    if (tid < 16) {
        const int j    = tid;
        const int lsel = (j < 8) ? 0 : 16;      // which half-wave holds row j
        const int r    = (j < 8) ? j : (j - 8);
        float b0 = -3.0e38f, b1 = -3.0e38f, b2 = -3.0e38f, b3 = -3.0e38f;
        int   j0 = 0, j1 = 0, j2 = 0, j3 = 0;
        for (int w = 0; w < 4; ++w)
            for (int l = 0; l < 16; ++l)
                for (int s = 0; s < 4; ++s)
                    top4_insert(cvls[w][lsel + l][r][s], cils[w][lsel + l][r][s],
                                b0, b1, b2, b3, j0, j1, j2, j3);
        const size_t base = ((size_t)(rowbase + j) * NCHUNK + blockIdx.y) * 4;
        pv[base + 0] = b0; pv[base + 1] = b1; pv[base + 2] = b2; pv[base + 3] = b3;
        pi[base + 0] = j0; pi[base + 1] = j1; pi[base + 2] = j2; pi[base + 3] = j3;
    }
}

// ---------------------------------------------------------------------------
// Kernel 4: per row merge 32 partial top4 -> final top4 -> softmax -> gather
// values rows -> ctx[row] (f32).  One block per row, 128 threads.
// ---------------------------------------------------------------------------
__global__ __launch_bounds__(128)
void finalize_kernel(const float* __restrict__ pv, const int* __restrict__ pi,
                     const float* __restrict__ values, float* __restrict__ ctx) {
    __shared__ float attn[4];
    __shared__ int   idx4[4];

    const int row = blockIdx.x;
    const int tid = threadIdx.x;

    if (tid == 0) {
        float b0 = -3.0e38f, b1 = -3.0e38f, b2 = -3.0e38f, b3 = -3.0e38f;
        int   j0 = 0, j1 = 0, j2 = 0, j3 = 0;
        const size_t base = (size_t)row * NCHUNK * 4;
        for (int c = 0; c < NCHUNK * 4; ++c)
            top4_insert(pv[base + c], pi[base + c], b0, b1, b2, b3, j0, j1, j2, j3);
        // softmax over top-4 (b0 is the max)
        float e0 = __expf(b0 - b0), e1 = __expf(b1 - b0);
        float e2 = __expf(b2 - b0), e3 = __expf(b3 - b0);
        float inv = 1.0f / (e0 + e1 + e2 + e3);
        attn[0] = e0 * inv; attn[1] = e1 * inv; attn[2] = e2 * inv; attn[3] = e3 * inv;
        idx4[0] = j0; idx4[1] = j1; idx4[2] = j2; idx4[3] = j3;
    }
    __syncthreads();

    #pragma unroll
    for (int i = 0; i < 4; ++i) {
        const int h = tid + i * 128;
        float a = 0.f;
        #pragma unroll
        for (int k = 0; k < 4; ++k)
            a += attn[k] * values[(size_t)idx4[k] * HDIM + h];
        ctx[(size_t)row * HDIM + h] = a;
    }
}

// ---------------------------------------------------------------------------
// Launch
// ---------------------------------------------------------------------------
extern "C" void kernel_launch(void* const* d_in, const int* in_sizes, int n_in,
                              void* d_out, int out_size, void* d_ws, size_t ws_size,
                              hipStream_t stream) {
    const float* query  = (const float*)d_in[0];
    const float* keys   = (const float*)d_in[1];
    const float* values = (const float*)d_in[2];
    const float* Wq     = (const float*)d_in[3];
    const float* Wk     = (const float*)d_in[4];
    const float* Wo     = (const float*)d_in[5];

    char* ws = (char*)d_ws;
    size_t off = 0;
    auto alloc = [&](size_t bytes) -> void* {
        void* p = ws + off;
        off = (off + bytes + 255) & ~(size_t)255;
        return p;
    };
    unsigned short* Kn  = (unsigned short*)alloc((size_t)NKEYS * HDIM * 2);  // 128 MB
    unsigned short* Qn  = (unsigned short*)alloc((size_t)BQ * HDIM * 2);     //   2 MB
    unsigned short* Wqb = (unsigned short*)alloc((size_t)HDIM * HDIM * 2);
    unsigned short* Wkb = (unsigned short*)alloc((size_t)HDIM * HDIM * 2);
    unsigned short* Wob = (unsigned short*)alloc((size_t)HDIM * HDIM * 2);
    float* pv  = (float*)alloc((size_t)BQ * NCHUNK * 4 * 4);
    int*   pi  = (int*)  alloc((size_t)BQ * NCHUNK * 4 * 4);
    float* ctx = (float*)alloc((size_t)BQ * HDIM * 4);

    const int wn = HDIM * HDIM;
    cvt_bf16_kernel<<<(wn + 255) / 256, 256, 0, stream>>>(Wq, Wqb, wn);
    cvt_bf16_kernel<<<(wn + 255) / 256, 256, 0, stream>>>(Wk, Wkb, wn);
    cvt_bf16_kernel<<<(wn + 255) / 256, 256, 0, stream>>>(Wo, Wob, wn);

    // Qn = normalize(query @ Wq.T) (bf16), Kn = normalize(keys @ Wk.T) (bf16)
    gemm_xwT_kernel<true><<<BQ / 16, 128, 0, stream>>>(query, Wqb, (void*)Qn);
    gemm_xwT_kernel<true><<<NKEYS / 16, 128, 0, stream>>>(keys, Wkb, (void*)Kn);

    // fused sim + top-4 partials
    sim_topk_kernel<<<dim3(BQ / 16, NCHUNK), 128, 0, stream>>>(Qn, Kn, pv, pi);

    // merge partials, softmax, gather values -> ctx
    finalize_kernel<<<BQ, 128, 0, stream>>>(pv, pi, values, ctx);

    // out = ctx @ Wo.T (f32)
    gemm_xwT_kernel<false><<<BQ / 16, 128, 0, stream>>>(ctx, Wob, d_out);
    (void)in_sizes; (void)n_in; (void)out_size; (void)ws_size;
}
